// SelfAttention_19576460935771
// MI455X (gfx1250) — matrix operand order repaired and checked by reading, not compiled, not run on hardware
//
#include <hip/hip_runtime.h>

// ---------------------------------------------------------------------------
// Problem constants (from reference): B=4, C=256, T=2048, H=8, D=32, G=32
// ---------------------------------------------------------------------------
#define BB 4
#define CC 256
#define TT 2048
#define HH 8
#define DD 32
#define GG 32
#define O3 (3 * CC)          // 768
#define SCALE 0.0625f        // 1/sqrt(256)

typedef __attribute__((ext_vector_type(16))) __bf16 v16bf;
typedef __attribute__((ext_vector_type(8)))  __bf16 v8bf;
typedef __attribute__((ext_vector_type(8)))  float  v8f;
typedef __attribute__((ext_vector_type(4)))  unsigned uint32x4;
typedef __attribute__((ext_vector_type(8)))  int      int32x8;
typedef __attribute__((ext_vector_type(4)))  int      int32x4;

// float -> bf16, round-to-nearest-even (bit trick)
__device__ __forceinline__ __bf16 f2bf(float f) {
    unsigned u = __builtin_bit_cast(unsigned, f);
    unsigned r = (u + 0x7FFFu + ((u >> 16) & 1u)) >> 16;
    unsigned short s = (unsigned short)r;
    return __builtin_bit_cast(__bf16, s);
}

// A-fragment (16x32 bf16, row-major source [m][k], stride in elements).
// lanes 0-15: row M=lane, K 0..15; lanes 16-31: row M=lane-16,
// K 16..31.
__device__ __forceinline__ v16bf load_a_frag(const __bf16* base, int stride, int lane) {
    int m = lane & 15, half = lane >> 4;
    const __bf16* p = base + (size_t)m * stride;
    union { v16bf v; v8bf h[2]; } u;
    u.v = *(const v16bf*)(p + 16 * half);
    return u.v;
}

// B-fragment (32x16 bf16) from N-major storage [n][k] (k contiguous).
// lanes 0-15: col N=lane, K=0..15; lanes 16-31: col N=lane-16, K=16..31.
__device__ __forceinline__ v16bf load_b_frag(const __bf16* base, int stride, int lane) {
    int n = lane & 15, half = lane >> 4;
    return *(const v16bf*)(base + (size_t)n * stride + 16 * half);
}

__device__ __forceinline__ v8f wmma_bf16(v16bf a, v16bf b, v8f c) {
    return __builtin_amdgcn_wmma_f32_16x16x32_bf16(
        /*neg_a=*/false, a, /*neg_b=*/false, b,
        /*c_mod=*/(short)0, c, /*reuse_a=*/false, /*reuse_b=*/false);
}

// Reductions across the 16 lanes of a wave32 half (N dim of a C fragment)
__device__ __forceinline__ float rowmax16(float v) {
    #pragma unroll
    for (int m = 1; m < 16; m <<= 1) v = fmaxf(v, __shfl_xor(v, m, 32));
    return v;
}
__device__ __forceinline__ float rowsum16(float v) {
    #pragma unroll
    for (int m = 1; m < 16; m <<= 1) v += __shfl_xor(v, m, 32);
    return v;
}

// ---------------------------------------------------------------------------
// Tensor Data Mover: 2-D tile Global -> LDS.  Builds D# groups per CDNA5 ISA
// (CDNA5 ISA) and issues the 6-arg builtin
// (uint32x4 g0, int32x8 g1, int32x4 g2, int32x4 g3, int32x8 pad, i32 cpol).
//   group0: count=1 | lds_addr | global_addr[56:0] | type=2
//   group1: data_size=2B, tensor_dim0/1, tile_dim0/1 (tile_dim2=0),
//           tensor_dim0_stride; groups 2/3 zero (2-D tensor).
// dim0 = contiguous (elements), dim1 = rows.
// ---------------------------------------------------------------------------
__device__ __forceinline__ void tdm_load_2d(const __bf16* gptr, unsigned lds_off,
                                            unsigned tensor_d0, unsigned tensor_d1,
                                            unsigned tile_d0, unsigned tile_d1,
                                            unsigned long long stride0) {
    unsigned long long ga = (unsigned long long)(uintptr_t)gptr;
    uint32x4 g0;
    g0[0] = 1u;                                              // count=1, user mode
    g0[1] = lds_off;                                         // lds_addr (bytes)
    g0[2] = (unsigned)(ga & 0xFFFFFFFFu);                    // global_addr[31:0]
    g0[3] = (unsigned)((ga >> 32) & 0x01FFFFFFu) | (2u << 30); // [56:32] | type=2
    int32x8 g1;
    g1[0] = (int)(1u << 16);                                 // data_size = 2 bytes
    g1[1] = (int)((tensor_d0 & 0xFFFFu) << 16);              // dim0[15:0] @ bit48
    g1[2] = (int)((tensor_d0 >> 16) | ((tensor_d1 & 0xFFFFu) << 16));
    g1[3] = (int)((tensor_d1 >> 16) | (tile_d0 << 16));      // tile_dim0 @ bit112
    g1[4] = (int)tile_d1;                                    // tile_dim1; tile_dim2=0
    g1[5] = (int)(unsigned)(stride0 & 0xFFFFFFFFu);          // dim0 stride low
    g1[6] = (int)(unsigned)((stride0 >> 32) & 0xFFFFu);      // dim0 stride high
    g1[7] = 0;
    int32x4 z4 = {0, 0, 0, 0};
    int32x8 z8 = {0, 0, 0, 0, 0, 0, 0, 0};
    __builtin_amdgcn_tensor_load_to_lds(g0, g1, z4, z4, z8, 0);
}

// ---------------------------------------------------------------------------
// Kernel 0: fp32 -> bf16 weight conversion
// ---------------------------------------------------------------------------
__global__ void convert_bf16_kernel(const float* __restrict__ in,
                                    __bf16* __restrict__ out, int n) {
    int i = blockIdx.x * 256 + threadIdx.x;
    if (i < n) out[i] = f2bf(in[i]);
}

// ---------------------------------------------------------------------------
// Kernel 1: GroupNorm over (C/G=8, T) per (b,g); writes norm as [B][T][C] bf16
// ---------------------------------------------------------------------------
__global__ void groupnorm_kernel(const float* __restrict__ x,
                                 const float* __restrict__ gn_w,
                                 const float* __restrict__ gn_b,
                                 __bf16* __restrict__ norm_t) {
    const int b = blockIdx.x / GG;
    const int g = blockIdx.x % GG;
    const int tid = threadIdx.x;
    const float* xb = x + (size_t)b * CC * TT + (size_t)g * 8 * TT;

    float s = 0.f, ss = 0.f;
    for (int i = tid; i < 8 * TT; i += 256) {
        float v = xb[i];
        s += v; ss += v * v;
    }
    __shared__ float rs[256], rq[256];
    rs[tid] = s; rq[tid] = ss;
    __syncthreads();
    for (int off = 128; off > 0; off >>= 1) {
        if (tid < off) { rs[tid] += rs[tid + off]; rq[tid] += rq[tid + off]; }
        __syncthreads();
    }
    const float mu  = rs[0] * (1.f / (8.f * TT));
    const float var = rq[0] * (1.f / (8.f * TT)) - mu * mu;
    const float inv = rsqrtf(var + 1e-5f);

    for (int i = tid; i < 8 * TT; i += 256) {
        int c = g * 8 + i / TT;
        int t = i % TT;
        float v = (xb[i] - mu) * inv * gn_w[c] + gn_b[c];
        norm_t[((size_t)b * TT + t) * CC + c] = f2bf(v);
    }
}

// ---------------------------------------------------------------------------
// Kernel 2: QKV GEMM  qkv[b][o][t] = sum_c w[o][c]*norm[b][c][t]
// A = qkv_w_bf16 [768][256], B = norm_t [B][T][C]; scatter to
// q,k: [B][H][T][D] and v: [B][H][D][T].
// ---------------------------------------------------------------------------
__global__ void qkv_gemm_kernel(const __bf16* __restrict__ w,
                                const __bf16* __restrict__ norm_t,
                                __bf16* __restrict__ qb,
                                __bf16* __restrict__ kb,
                                __bf16* __restrict__ vb) {
    const int b    = blockIdx.z;
    const int lane = threadIdx.x & 31;
    const int wid  = threadIdx.x >> 5;
    const int o0   = blockIdx.y * 64 + 16 * (wid & 3);
    const int t0   = blockIdx.x * 64 + 32 * (wid >> 2);
    const __bf16* nb = norm_t + (size_t)b * TT * CC;

    v8f c0 = {}, c1 = {};
    #pragma unroll
    for (int k0 = 0; k0 < CC; k0 += 32) {
        v16bf a  = load_a_frag(w + (size_t)o0 * CC + k0, CC, lane);
        v16bf b0 = load_b_frag(nb + (size_t)t0 * CC + k0, CC, lane);
        v16bf b1 = load_b_frag(nb + (size_t)(t0 + 16) * CC + k0, CC, lane);
        c0 = wmma_bf16(a, b0, c0);
        c1 = wmma_bf16(a, b1, c1);
    }
    const int half = lane >> 4, n = lane & 15;
    #pragma unroll
    for (int i = 0; i < 8; i++) {
        int o = o0 + i + 8 * half;
        int h = o / 96, r = o % 96;
        size_t bh = (size_t)(b * HH + h);
        int ta = t0 + n, tb2 = t0 + 16 + n;
        if (r < 32) {
            qb[(bh * TT + ta)  * DD + r] = f2bf(c0[i]);
            qb[(bh * TT + tb2) * DD + r] = f2bf(c1[i]);
        } else if (r < 64) {
            kb[(bh * TT + ta)  * DD + (r - 32)] = f2bf(c0[i]);
            kb[(bh * TT + tb2) * DD + (r - 32)] = f2bf(c1[i]);
        } else {
            vb[(bh * DD + (r - 64)) * TT + ta ] = f2bf(c0[i]);
            vb[(bh * DD + (r - 64)) * TT + tb2] = f2bf(c1[i]);
        }
    }
}

// ---------------------------------------------------------------------------
// Kernel 3: flash attention per (b,h). Wave = 16 query rows; block = 8 waves
// = 128 queries. 64-key tiles are staged Global->LDS by the Tensor Data
// Mover (wave 0 issues tensor_load_to_lds, waits TENSORcnt, block barrier),
// so the 8 waves share one copy instead of 8 redundant global streams.
// Online softmax; O kept in C fragments. Output: attn_t [B][T][C] bf16.
// ---------------------------------------------------------------------------
__global__ void attention_kernel(const __bf16* __restrict__ qb,
                                 const __bf16* __restrict__ kb,
                                 const __bf16* __restrict__ vb,
                                 __bf16* __restrict__ attn_t) {
    const int b = blockIdx.z, h = blockIdx.y;
    const int lane = threadIdx.x & 31;
    const int wv   = threadIdx.x >> 5;
    const int q0   = blockIdx.x * 128 + wv * 16;
    const size_t bh = (size_t)(b * HH + h);
    const __bf16* qptr = qb + bh * TT * DD;
    const __bf16* kptr = kb + bh * TT * DD;
    const __bf16* vptr = vb + bh * DD * TT;

    __shared__ __align__(128) __bf16 lds_k[64 * DD];   // [key][d]   4KB
    __shared__ __align__(128) __bf16 lds_v[DD * 64];   // [d][key]   4KB
    __shared__ __align__(32)  __bf16 pt[8][16 * 32];   // P bounce   8KB

    const v16bf qa = load_a_frag(qptr + (size_t)q0 * DD, DD, lane);

    v8f o0 = {}, o1 = {};
    float mrow[8], lrow[8];
    #pragma unroll
    for (int i = 0; i < 8; i++) { mrow[i] = -3.0e38f; lrow[i] = 0.f; }
    const v8f zero = {};
    const int half = lane >> 4, nn = lane & 15;

    for (int tko = 0; tko < TT; tko += 64) {
        __syncthreads();   // previous tile fully consumed
        if (wv == 0) {
            // K tile: 64 rows of 32 contiguous bf16 (row stride DD)
            tdm_load_2d(kptr + (size_t)tko * DD,
                        (unsigned)(uintptr_t)&lds_k[0],
                        DD, TT, DD, 64, DD);
            // V tile: 32 rows of 64 contiguous bf16 (row stride TT)
            tdm_load_2d(vptr + tko,
                        (unsigned)(uintptr_t)&lds_v[0],
                        TT, DD, 64, DD, TT);
            __builtin_amdgcn_s_wait_tensorcnt(0);
        }
        __syncthreads();   // tile visible to all waves

        #pragma unroll
        for (int sub = 0; sub < 64; sub += 32) {
            v16bf kf0 = load_b_frag(&lds_k[(size_t)sub * DD], DD, lane);
            v16bf kf1 = load_b_frag(&lds_k[(size_t)(sub + 16) * DD], DD, lane);
            v8f s0 = wmma_bf16(qa, kf0, zero);
            v8f s1 = wmma_bf16(qa, kf1, zero);

            #pragma unroll
            for (int i = 0; i < 8; i++) {
                float a  = s0[i] * SCALE;
                float c  = s1[i] * SCALE;
                float rm = rowmax16(fmaxf(a, c));
                float mn = fmaxf(mrow[i], rm);
                float alpha = __expf(mrow[i] - mn);
                float p0 = __expf(a - mn);
                float p1 = __expf(c - mn);
                float rsum = rowsum16(p0 + p1);
                lrow[i] = lrow[i] * alpha + rsum;
                mrow[i] = mn;
                o0[i] *= alpha;
                o1[i] *= alpha;
                int m = i + 8 * half;
                pt[wv][m * 32 + nn]      = f2bf(p0);
                pt[wv][m * 32 + 16 + nn] = f2bf(p1);
            }
            // P: C-layout f32 -> A-layout bf16 via LDS (in-order DS pipe)
            v16bf pa  = load_a_frag(&pt[wv][0], 32, lane);
            v16bf vf0 = load_b_frag(&lds_v[sub], 64, lane);            // d 0..15
            v16bf vf1 = load_b_frag(&lds_v[16 * 64 + sub], 64, lane);  // d 16..31
            o0 = wmma_bf16(pa, vf0, o0);
            o1 = wmma_bf16(pa, vf1, o1);
        }
    }

    #pragma unroll
    for (int i = 0; i < 8; i++) {
        float invl = 1.f / lrow[i];
        int t = q0 + i + 8 * half;
        __bf16* row = attn_t + ((size_t)b * TT + t) * CC + h * DD;
        row[nn]      = f2bf(o0[i] * invl);
        row[16 + nn] = f2bf(o1[i] * invl);
    }
}

// ---------------------------------------------------------------------------
// Kernel 4: out-projection + bias + residual (fp32 out).
// A = out_w_bf16 [256][256], B = attn_t [B][T][C]; d_out = [B][C][T] fp32.
// ---------------------------------------------------------------------------
__global__ void outproj_kernel(const __bf16* __restrict__ w,
                               const __bf16* __restrict__ attn_t,
                               const float* __restrict__ out_b,
                               const float* __restrict__ x,
                               float* __restrict__ out) {
    const int b    = blockIdx.z;
    const int lane = threadIdx.x & 31;
    const int wid  = threadIdx.x >> 5;
    const int o0   = blockIdx.y * 64 + 16 * (wid & 3);
    const int t0   = blockIdx.x * 64 + 32 * (wid >> 2);
    const __bf16* ab = attn_t + (size_t)b * TT * CC;

    v8f c0 = {}, c1 = {};
    #pragma unroll
    for (int k0 = 0; k0 < CC; k0 += 32) {
        v16bf a  = load_a_frag(w + (size_t)o0 * CC + k0, CC, lane);
        v16bf b0 = load_b_frag(ab + (size_t)t0 * CC + k0, CC, lane);
        v16bf b1 = load_b_frag(ab + (size_t)(t0 + 16) * CC + k0, CC, lane);
        c0 = wmma_bf16(a, b0, c0);
        c1 = wmma_bf16(a, b1, c1);
    }
    const int half = lane >> 4, n = lane & 15;
    #pragma unroll
    for (int i = 0; i < 8; i++) {
        int o = o0 + i + 8 * half;
        float bias = out_b[o];
        size_t base = ((size_t)b * CC + o) * TT;
        int ta = t0 + n, tb2 = t0 + 16 + n;
        out[base + ta]  = c0[i] + bias + x[base + ta];
        out[base + tb2] = c1[i] + bias + x[base + tb2];
    }
}

// ---------------------------------------------------------------------------
extern "C" void kernel_launch(void* const* d_in, const int* in_sizes, int n_in,
                              void* d_out, int out_size, void* d_ws, size_t ws_size,
                              hipStream_t stream) {
    const float* x     = (const float*)d_in[0];
    const float* gn_w  = (const float*)d_in[1];
    const float* gn_b  = (const float*)d_in[2];
    const float* qkv_w = (const float*)d_in[3];
    const float* out_w = (const float*)d_in[4];
    const float* out_b = (const float*)d_in[5];
    float* out = (float*)d_out;

    // Workspace carving (~20.5 MB total)
    char* ws = (char*)d_ws;
    __bf16* norm_t = (__bf16*)ws;  ws += (size_t)BB * TT * CC * 2;        // 4 MB
    __bf16* qkvw_b = (__bf16*)ws;  ws += (size_t)O3 * CC * 2;             // 384 KB
    __bf16* outw_b = (__bf16*)ws;  ws += (size_t)CC * CC * 2;             // 128 KB
    __bf16* q_buf  = (__bf16*)ws;  ws += (size_t)BB * HH * TT * DD * 2;   // 4 MB
    __bf16* k_buf  = (__bf16*)ws;  ws += (size_t)BB * HH * TT * DD * 2;   // 4 MB
    __bf16* v_buf  = (__bf16*)ws;  ws += (size_t)BB * HH * DD * TT * 2;   // 4 MB
    __bf16* attn_t = (__bf16*)ws;  ws += (size_t)BB * TT * CC * 2;        // 4 MB

    // 0) weights fp32 -> bf16
    convert_bf16_kernel<<<(O3 * CC + 255) / 256, 256, 0, stream>>>(qkv_w, qkvw_b, O3 * CC);
    convert_bf16_kernel<<<(CC * CC + 255) / 256, 256, 0, stream>>>(out_w, outw_b, CC * CC);

    // 1) GroupNorm -> norm_t [B][T][C]
    groupnorm_kernel<<<BB * GG, 256, 0, stream>>>(x, gn_w, gn_b, norm_t);

    // 2) QKV GEMM -> q,k [B][H][T][D], v [B][H][D][T]
    qkv_gemm_kernel<<<dim3(TT / 64, O3 / 64, BB), 256, 0, stream>>>(
        qkvw_b, norm_t, q_buf, k_buf, v_buf);

    // 3) Flash attention (TDM-staged K/V tiles) -> attn_t [B][T][C]
    attention_kernel<<<dim3(TT / 128, HH, BB), 256, 0, stream>>>(
        q_buf, k_buf, v_buf, attn_t);

    // 4) Out projection + bias + residual -> d_out [B][C][T] fp32
    outproj_kernel<<<dim3(TT / 64, CC / 64, BB), 256, 0, stream>>>(
        outw_b, attn_t, out_b, x, out);
}
